// srcnn_39187281608754
// MI455X (gfx1250) — 16-bit and 8-bit matrix operand order repaired and checked by reading, compiled, not run on hardware
//
#include <hip/hip_runtime.h>
#include <hip/hip_bf16.h>

typedef __attribute__((ext_vector_type(16))) _Float16     v16h;
typedef __attribute__((ext_vector_type(8)))  float        v8f;
typedef __attribute__((ext_vector_type(8)))  int          v8i;
typedef __attribute__((ext_vector_type(4)))  unsigned int v4u;

#define BATCH 8
#define HW    512
#define C1    64
#define C2    32

#if __has_builtin(__builtin_amdgcn_global_load_async_to_lds_b128) && \
    __has_builtin(__builtin_amdgcn_s_wait_asynccnt)
#define HAVE_ASYNC_LDS 1
#else
#define HAVE_ASYNC_LDS 0
#endif
#define AS1 __attribute__((address_space(1)))
#define AS3 __attribute__((address_space(3)))
typedef int async16_t __attribute__((vector_size(16)));

#define WS_H1    0ULL
#define WS_H2    (WS_H1 + (size_t)BATCH*HW*HW*C1)
#define WS_QW1F  (WS_H2 + (size_t)BATCH*HW*HW*C2)
#define WS_QW2F  (WS_QW1F + (size_t)4*3*32*16*2)
#define WS_QW3F  (WS_QW2F + (size_t)2*25*32*32)
#define WS_SCAL  (WS_QW3F + (size_t)13*32*32)

__device__ inline float block_max_abs(const float* w, int n, float* red) {
  float m = 0.f;
  for (int i = threadIdx.x; i < n; i += 256) m = fmaxf(m, fabsf(w[i]));
  red[threadIdx.x] = m; __syncthreads();
  for (int s = 128; s > 0; s >>= 1) {
    if ((int)threadIdx.x < s) red[threadIdx.x] = fmaxf(red[threadIdx.x], red[threadIdx.x + s]);
    __syncthreads();
  }
  return red[0];
}
__device__ inline int q8i(float w, float inv_s) {
  float q = rintf(w * inv_s);
  return (int)fminf(fmaxf(q, -127.f), 127.f);
}

__global__ __launch_bounds__(256) void k_prep(const float* __restrict__ w1,
                                              const float* __restrict__ w2,
                                              const float* __restrict__ w3,
                                              unsigned char* __restrict__ ws) {
  __shared__ float red[256];
  _Float16*    qw1f = (_Float16*)(ws + WS_QW1F);
  signed char* qw2f = (signed char*)(ws + WS_QW2F);
  signed char* qw3f = (signed char*)(ws + WS_QW3F);
  float*       scal = (float*)(ws + WS_SCAL);

  if (blockIdx.x == 0) {
    float mx = block_max_abs(w1, 64 * 81, red);
    float inv = 127.f / mx;
    if (threadIdx.x == 0) scal[0] = mx / 127.f;
    for (int e = threadIdx.x; e < 4 * 3 * 32 * 16; e += 256) {
      int i = e & 15, l = (e >> 4) & 31, ks = (e >> 9) % 3, t = e / (3 * 32 * 16);
      int hi = l >> 4;
      int k = ks * 32 + i + (hi ? 16 : 0);
      int m = t * 16 + (l & 15);
      float v = (k < 81) ? (float)q8i(w1[m * 81 + k], inv) : 0.f;
      qw1f[e] = (_Float16)v;
    }
  } else if (blockIdx.x == 1) {
    float mx = block_max_abs(w2, 32 * 64 * 25, red);
    float inv = 127.f / mx;
    if (threadIdx.x == 0) scal[1] = mx / 127.f;
    for (int e = threadIdx.x; e < 2 * 25 * 32 * 32; e += 256) {
      int i = e & 31, l = (e >> 5) & 31, tap = (e >> 10) % 25, t = e / (25 * 32 * 32);
      int hi = l >> 4;
      int kch = (i >> 4) * 32 + (i & 15) + (hi ? 16 : 0);
      int m = t * 16 + (l & 15), ty = tap / 5, tx = tap % 5;
      qw2f[e] = (signed char)q8i(w2[((m * 64 + kch) * 5 + ty) * 5 + tx], inv);
    }
  } else {
    float mx = block_max_abs(w3, 800, red);
    float inv = 127.f / mx;
    if (threadIdx.x == 0) scal[2] = mx / 127.f;
    for (int e = threadIdx.x; e < 13 * 32 * 32; e += 256) {
      int i = e & 31, l = (e >> 5) & 31, c = e >> 10;
      int hi = l >> 4;
      int k = (i >> 4) * 32 + (i & 15) + (hi ? 16 : 0);
      int tap = 2 * c + (k >= 32), ch = k & 31;
      signed char v = 0;
      if ((l & 15) == 0 && tap < 25) {
        int ty = tap / 5, tx = tap % 5;
        v = (signed char)q8i(w3[(ch * 5 + ty) * 5 + tx], inv);
      }
      qw3f[e] = v;
    }
  }
}

__global__ __launch_bounds__(256) void k_conv1(const float* __restrict__ x,
                                               const float* __restrict__ b1,
                                               const float* __restrict__ s1p,
                                               unsigned char* __restrict__ ws) {
  __shared__ __align__(32) _Float16 patches[128 * 96];
  const int tid = threadIdx.x;
  const int b = blockIdx.z, y = blockIdx.y, xbase = blockIdx.x * 128;
  const float* xin = x + (size_t)b * HW * HW;

  for (int e = tid; e < 128 * 96; e += 256) {
    int px = e / 96, k = e % 96;
    float v = 0.f;
    if (k < 81) {
      int ry = k / 9, rx = k % 9;
      int gy = y + ry - 4, gx = xbase + px + rx - 4;
      if ((unsigned)gy < HW && (unsigned)gx < HW) v = xin[(size_t)gy * HW + gx];
    }
    patches[e] = (_Float16)v;
  }
  __syncthreads();

  const _Float16* qw1f = (const _Float16*)(ws + WS_QW1F);
  const float*    scal = (const float*)(ws + WS_SCAL);
  unsigned char*  h1   = ws + WS_H1;

  const int lane = tid & 31, wv = tid >> 5;
  const int col = lane & 15, hi = lane >> 4;
  const int pcol = wv * 16 + col;

  v8f acc[4] = {{}, {}, {}, {}};
  #pragma unroll
  for (int ks = 0; ks < 3; ++ks) {
    v16h bf = *(const v16h*)(&patches[pcol * 96 + ks * 32 + hi * 16]);
    #pragma unroll
    for (int t = 0; t < 4; ++t) {
      v16h af = *(const v16h*)(qw1f + ((t * 3 + ks) * 32 + lane) * 16);
      acc[t] = __builtin_amdgcn_wmma_f32_16x16x32_f16(false, af, false, bf,
                                                      (short)0, acc[t], false, false);
    }
  }

  const float sw1 = scal[0];
  const float s1 = s1p[0], istep1 = 255.f / s1;
  const int px = xbase + pcol;
  const size_t obase = ((((size_t)b * HW + y) * HW) + px) * C1 + hi * 8;
  #pragma unroll
  for (int t = 0; t < 4; ++t) {
    unsigned int p0 = 0, p1 = 0;
    #pragma unroll
    for (int r = 0; r < 8; ++r) {
      int ch = t * 16 + hi * 8 + r;
      float v = acc[t][r] * sw1 + b1[ch];
      float q = rintf(fminf(fmaxf(v, 0.f), s1) * istep1);
      unsigned int qi = (unsigned int)(int)fminf(q, 255.f);
      if (r < 4) p0 |= qi << (8 * r); else p1 |= qi << (8 * (r - 4));
    }
    *(uint2*)(h1 + obase + t * 16) = make_uint2(p0, p1);
  }
}

__device__ inline void fill_tile_async(unsigned char* tile, const unsigned char* src,
                                       int ch16_per_px, int b, int y, int xbase,
                                       int cbytes) {
  const int nchunk = 5 * 132 * ch16_per_px;
  for (int d = threadIdx.x; d < nchunk; d += 256) {
    int c16 = d % ch16_per_px;
    int pxl = (d / ch16_per_px) % 132;
    int row = d / (132 * ch16_per_px);
    int gy = y + row - 2, gx = xbase + pxl - 2;
    if ((unsigned)gy < HW && (unsigned)gx < HW) {
      size_t goff = ((((size_t)b * HW + gy) * HW + gx) * cbytes) + (size_t)c16 * 16;
#if HAVE_ASYNC_LDS
      __builtin_amdgcn_global_load_async_to_lds_b128(
          (AS1 async16_t*)(uintptr_t)(src + goff),
          (AS3 async16_t*)(uintptr_t)(tile + (size_t)d * 16), 0, 0);
#else
      *(v4u*)(tile + (size_t)d * 16) = *(const v4u*)(src + goff);
#endif
    } else {
      v4u z = {0, 0, 0, 0};
      *(v4u*)(tile + (size_t)d * 16) = z;
    }
  }
#if HAVE_ASYNC_LDS
  __builtin_amdgcn_s_wait_asynccnt(0);
#endif
  __syncthreads();
}

__global__ __launch_bounds__(256) void k_conv2(const float* __restrict__ b2,
                                               const float* __restrict__ s1p,
                                               const float* __restrict__ s2p,
                                               unsigned char* __restrict__ ws) {
  __shared__ __align__(16) unsigned char tile[5 * 132 * 64];
  const int tid = threadIdx.x;
  const int b = blockIdx.z, y = blockIdx.y, xbase = blockIdx.x * 128;

  fill_tile_async(tile, ws + WS_H1,  4, b, y, xbase, C1);

  const v8i*   afrag = (const v8i*)(ws + WS_QW2F);
  const float* scal  = (const float*)(ws + WS_SCAL);
  unsigned char* h2  = ws + WS_H2;

  const int lane = tid & 31, wv = tid >> 5;
  const int col = lane & 15, hi = lane >> 4;
  const int col0 = wv * 16;

  v8i acc0 = {}, acc1 = {};
  #pragma unroll
  for (int ty = 0; ty < 5; ++ty) {
    #pragma unroll
    for (int tx = 0; tx < 5; ++tx) {
      const unsigned char* p = tile + ((ty * 132 + (col0 + col + tx)) * 64) + hi * 16;
      v4u lo = *(const v4u*)p;
      v4u hh = *(const v4u*)(p + 32);
      v8i bf;
      bf[0] = (int)lo[0]; bf[1] = (int)lo[1]; bf[2] = (int)lo[2]; bf[3] = (int)lo[3];
      bf[4] = (int)hh[0]; bf[5] = (int)hh[1]; bf[6] = (int)hh[2]; bf[7] = (int)hh[3];
      int tap = ty * 5 + tx;
      v8i a0 = afrag[(0 * 25 + tap) * 32 + lane];
      v8i a1 = afrag[(1 * 25 + tap) * 32 + lane];
      acc0 = __builtin_amdgcn_wmma_i32_16x16x64_iu8(true, a0, false, bf, acc0, false, false);
      acc1 = __builtin_amdgcn_wmma_i32_16x16x64_iu8(true, a1, false, bf, acc1, false, false);
    }
  }

  const float kk = (s1p[0] / 255.f) * scal[1];
  const float s2 = s2p[0], istep2 = 255.f / s2;
  const int px = xbase + col0 + col;
  const size_t obase = ((((size_t)b * HW + y) * HW) + px) * C2 + hi * 8;
  #pragma unroll
  for (int t = 0; t < 2; ++t) {
    v8i a = t ? acc1 : acc0;
    unsigned int p0 = 0, p1 = 0;
    #pragma unroll
    for (int r = 0; r < 8; ++r) {
      float v = (float)a[r] * kk + b2[t * 16 + hi * 8 + r];
      float q = rintf(fminf(fmaxf(v, 0.f), s2) * istep2);
      unsigned int qi = (unsigned int)(int)fminf(q, 255.f);
      if (r < 4) p0 |= qi << (8 * r); else p1 |= qi << (8 * (r - 4));
    }
    *(uint2*)(h2 + obase + t * 16) = make_uint2(p0, p1);
  }
}

__global__ __launch_bounds__(256) void k_conv3(const float* __restrict__ b3,
                                               const float* __restrict__ s2p,
                                               const float* __restrict__ s3p,
                                               unsigned char* __restrict__ ws,
                                               float* __restrict__ out) {
  __shared__ __align__(16) unsigned char tile[5 * 132 * 32];
  const int tid = threadIdx.x;
  const int b = blockIdx.z, y = blockIdx.y, xbase = blockIdx.x * 128;

  fill_tile_async(tile, ws + WS_H2,  2, b, y, xbase, C2);

  const v8i*   afrag = (const v8i*)(ws + WS_QW3F);
  const float* scal  = (const float*)(ws + WS_SCAL);

  const int lane = tid & 31, wv = tid >> 5;
  const int col = lane & 15, hi = lane >> 4;
  const int col0 = wv * 16;

  v8i acc = {};
  #pragma unroll
  for (int c = 0; c < 13; ++c) {
    int tapA = 2 * c, tapB = 2 * c + 1;
    const unsigned char* pA =
        tile + ((tapA / 5 * 132 + (col0 + col + tapA % 5)) * 32) + hi * 16;
    v4u lo = *(const v4u*)pA;
    v4u hh = {0, 0, 0, 0};
    if (tapB < 25) {
      const unsigned char* pB =
          tile + ((tapB / 5 * 132 + (col0 + col + tapB % 5)) * 32) + hi * 16;
      hh = *(const v4u*)pB;
    }
    v8i bf;
    bf[0] = (int)lo[0]; bf[1] = (int)lo[1]; bf[2] = (int)lo[2]; bf[3] = (int)lo[3];
    bf[4] = (int)hh[0]; bf[5] = (int)hh[1]; bf[6] = (int)hh[2]; bf[7] = (int)hh[3];
    v8i a = afrag[c * 32 + lane];
    acc = __builtin_amdgcn_wmma_i32_16x16x64_iu8(true, a, false, bf, acc, false, false);
  }

  if (hi == 0) {
    const float kk = (s2p[0] / 255.f) * scal[2];
    const float s3 = s3p[0], step3 = s3 / 255.f;
    float v = (float)acc[0] * kk + b3[0];
    float q = rintf(fminf(fmaxf(v, 0.f), s3) * (255.f / s3));
    out[(((size_t)b * HW) + y) * HW + xbase + col0 + col] = q * step3;
  }
}

extern "C" void kernel_launch(void* const* d_in, const int* in_sizes, int n_in,
                              void* d_out, int out_size, void* d_ws, size_t ws_size,
                              hipStream_t stream) {
  const float* x  = (const float*)d_in[0];
  const float* w1 = (const float*)d_in[1];
  const float* b1 = (const float*)d_in[2];
  const float* w2 = (const float*)d_in[3];
  const float* b2 = (const float*)d_in[4];
  const float* w3 = (const float*)d_in[5];
  const float* b3 = (const float*)d_in[6];
  const float* s1 = (const float*)d_in[7];
  const float* s2 = (const float*)d_in[8];
  const float* s3 = (const float*)d_in[9];
  unsigned char* ws = (unsigned char*)d_ws;
  float* out = (float*)d_out;

  k_prep<<<3, 256, 0, stream>>>(w1, w2, w3, ws);
  dim3 grid(HW / 128, HW, BATCH);
  k_conv1<<<grid, 256, 0, stream>>>(x, b1, s1, ws);
  k_conv2<<<grid, 256, 0, stream>>>(b2, s1, s2, ws);
  k_conv3<<<grid, 256, 0, stream>>>(b3, s2, s3, ws, out);
}
